// GNNEncoder_86947317940720
// MI455X (gfx1250) — matrix operand order repaired and checked by reading, not compiled, not run on hardware
//
#include <hip/hip_runtime.h>
#include <hip/hip_bf16.h>

typedef __attribute__((ext_vector_type(16))) _Float16 v16h;
typedef __attribute__((ext_vector_type(8)))  float    v8f;

#define D 128
#define TILE_M 64          // node rows per block (4 x 16-row WMMA tiles)

// ---------------------------------------------------------------------------
// Zero a float buffer.
__global__ __launch_bounds__(256) void sage_zero_f32(float* __restrict__ p, int n) {
    int i = blockIdx.x * 256 + threadIdx.x;
    if (i < n) p[i] = 0.0f;
}

// ---------------------------------------------------------------------------
// Pack a 128x128 f32 weight matrix (stored [K,N] row-major) into the exact
// CDNA5 WMMA B-fragment order for V_WMMA_F32_16X16X32_F16:
//   B 32x16 f16 tile: lanes 0-15 hold K=0..7,16..23 (col = lane), lanes 16-31 hold
//   K=8..15,24..31 (col = lane-16). Per lane: 16 contiguous halves = one 32B load.
// packed index = ((ktile*8 + ntile)*32 + lane)*16 + i
__global__ __launch_bounds__(256) void sage_pack_w(const float* __restrict__ w,
                                                   _Float16* __restrict__ p) {
    int idx = blockIdx.x * 256 + threadIdx.x;
    if (idx >= D * D) return;
    int i     = idx & 15;
    int lane  = (idx >> 4) & 31;
    int t     = idx >> 9;          // 0..31
    int ntile = t & 7;
    int ktile = t >> 3;
    int k = ktile * 32 + 8 * (lane >> 4) + (i & 7) + 16 * (i >> 3);
    int n = ntile * 16 + (lane & 15);
    p[idx] = (_Float16)w[k * D + n];
}

// ---------------------------------------------------------------------------
// Degree: one thread per edge, hardware f32 atomic add.
__global__ __launch_bounds__(256) void sage_degree(const long long* __restrict__ dst,
                                                   float* __restrict__ deg, int E) {
    int e = blockIdx.x * 256 + threadIdx.x;
    if (e < E) {
        long long d = dst[e];
        __hip_atomic_fetch_add(&deg[d], 1.0f, __ATOMIC_RELAXED,
                               __HIP_MEMORY_SCOPE_AGENT);
    }
}

// deg -> 1/max(deg,1) in place (zero-degree rows have agg==0 anyway).
__global__ __launch_bounds__(256) void sage_invdeg(float* __restrict__ deg, int n) {
    int i = blockIdx.x * 256 + threadIdx.x;
    if (i < n) deg[i] = 1.0f / fmaxf(deg[i], 1.0f);
}

// ---------------------------------------------------------------------------
// Scatter-add: one wave32 per edge; lane handles 4 contiguous floats.
// Coalesced 512B source-row read, hardware global_atomic_add_f32 into agg.
__global__ __launch_bounds__(256) void sage_scatter(const float* __restrict__ h,
                                                    const long long* __restrict__ ei,
                                                    float* __restrict__ agg, int E) {
    int edge = blockIdx.x * 8 + (threadIdx.x >> 5);
    if (edge >= E) return;
    int lane = threadIdx.x & 31;
    long long src = ei[edge];
    long long dst = ei[E + edge];
    const float4 v = *(const float4*)(h + src * D + lane * 4);
    float* a = agg + dst * D + lane * 4;
    __hip_atomic_fetch_add(a + 0, v.x, __ATOMIC_RELAXED, __HIP_MEMORY_SCOPE_AGENT);
    __hip_atomic_fetch_add(a + 1, v.y, __ATOMIC_RELAXED, __HIP_MEMORY_SCOPE_AGENT);
    __hip_atomic_fetch_add(a + 2, v.z, __ATOMIC_RELAXED, __HIP_MEMORY_SCOPE_AGENT);
    __hip_atomic_fetch_add(a + 3, v.w, __ATOMIC_RELAXED, __HIP_MEMORY_SCOPE_AGENT);
}

// ---------------------------------------------------------------------------
// Fused SAGE layer GEMM:  out = mean @ Wl + h @ Wr + b  (+ optional ReLU)
// Block = 256 threads = 8 waves; block owns 64 node rows (4 M-tiles); wave w
// owns column tile ntile=w with 4 accumulator tiles. Each B fragment loaded
// once per K-step is reused across 4 A fragments: 32 WMMAs per wave.
__global__ __launch_bounds__(256) void sage_gemm(const float* __restrict__ h,
                                                 const float* __restrict__ agg,
                                                 const float* __restrict__ invdeg,
                                                 const _Float16* __restrict__ wlp,
                                                 const _Float16* __restrict__ wrp,
                                                 const float* __restrict__ bias,
                                                 float* __restrict__ out,
                                                 int n_nodes, int relu) {
    __shared__ _Float16 sH[TILE_M * D];   // h panel, f16      (16 KB)
    __shared__ _Float16 sM[TILE_M * D];   // mean panel, f16   (16 KB)

    const int tid  = threadIdx.x;
    const int row0 = blockIdx.x * TILE_M;

    // Stage 64x128 panels of h and mean (=agg*invdeg) into LDS as f16.
#pragma unroll
    for (int t = 0; t < 32; ++t) {
        int e = tid * 32 + t;             // 0..8191, 64B/thread LDS chunks
        int r = e >> 7;
        int c = e & (D - 1);
        int gr = row0 + r;
        if (gr >= n_nodes) gr = n_nodes - 1;   // clamp reads for tail block
        int g = gr * D + c;
        sH[e] = (_Float16)h[g];
        sM[e] = (_Float16)(agg[g] * invdeg[gr]);
    }
    __syncthreads();

    const int lane = tid & 31;
    const int wid  = tid >> 5;            // ntile 0..7
    const int arow = lane & 15;           // A-matrix row within 16-row tile
    const int hs   = lane >> 4;           // half-select (K grouping)

    v8f acc[4] = {};                      // 4 M-tiles per wave

#pragma unroll
    for (int kt = 0; kt < 4; ++kt) {
        const int k0 = kt * 32;
        // B fragments: contiguous 32B per lane from pre-packed weights.
        const int bo = ((kt * 8 + wid) * 32 + lane) * 16;
        const v16h bl = *(const v16h*)(wlp + bo);
        const v16h br = *(const v16h*)(wrp + bo);

#pragma unroll
        for (int mt = 0; mt < 4; ++mt) {
            // A fragment, 16-bit A 16x32 layout: two contiguous 8-half runs.
            //   lanes 0-15: K = k0+[0..7]  and k0+16+[0..7]
            //   lanes16-31: K = k0+8+[0..7] and k0+24+[0..7]
            const _Float16* mrow = sM + (mt * 16 + arow) * D;
            const _Float16* hrow = sH + (mt * 16 + arow) * D;
            const _Float16* p1m = mrow + k0 + 8 * hs;
            const _Float16* p1h = hrow + k0 + 8 * hs;
            v16h am, ah;
#pragma unroll
            for (int i = 0; i < 8; ++i) {
                am[i]     = p1m[i];
                am[8 + i] = p1m[16 + i];
                ah[i]     = p1h[i];
                ah[8 + i] = p1h[16 + i];
            }
            acc[mt] = __builtin_amdgcn_wmma_f32_16x16x32_f16(
                false, am, false, bl, (short)0, acc[mt], false, false);
            acc[mt] = __builtin_amdgcn_wmma_f32_16x16x32_f16(
                false, ah, false, br, (short)0, acc[mt], false, false);
        }
    }

    // C/D layout: VGPR r -> row (r + 8*hs), col = lane&15 within the tile.
    const int col = wid * 16 + (lane & 15);
    const float bv = bias[col];
#pragma unroll
    for (int mt = 0; mt < 4; ++mt) {
#pragma unroll
        for (int r = 0; r < 8; ++r) {
            int m = row0 + mt * 16 + r + 8 * hs;
            if (m < n_nodes) {
                float v = acc[mt][r] + bv;
                if (relu) v = fmaxf(v, 0.0f);
                out[m * D + col] = v;
            }
        }
    }
}

// ---------------------------------------------------------------------------
static inline size_t align256(size_t x) { return (x + 255) & ~size_t(255); }

extern "C" void kernel_launch(void* const* d_in, const int* in_sizes, int n_in,
                              void* d_out, int out_size, void* d_ws, size_t ws_size,
                              hipStream_t stream) {
    const float*     x  = (const float*)d_in[0];
    const long long* ei = (const long long*)d_in[1];
    const float* wl[3] = { (const float*)d_in[2], (const float*)d_in[5], (const float*)d_in[8] };
    const float* wr[3] = { (const float*)d_in[3], (const float*)d_in[6], (const float*)d_in[9] };
    const float* bb[3] = { (const float*)d_in[4], (const float*)d_in[7], (const float*)d_in[10] };

    const int N = in_sizes[0] / D;       // 50000
    const int E = in_sizes[1] / 2;       // 800000

    // Workspace carve-up.
    char* ws = (char*)d_ws;
    size_t off = 0;
    float* h1  = (float*)(ws + off); off = align256(off + (size_t)N * D * sizeof(float));
    float* h2  = (float*)(ws + off); off = align256(off + (size_t)N * D * sizeof(float));
    float* agg = (float*)(ws + off); off = align256(off + (size_t)N * D * sizeof(float));
    float* deg = (float*)(ws + off); off = align256(off + (size_t)N * sizeof(float));
    _Float16* pw[6];
    for (int i = 0; i < 6; ++i) {
        pw[i] = (_Float16*)(ws + off);
        off = align256(off + (size_t)D * D * sizeof(_Float16));
    }
    (void)ws_size;

    const int ND = N * D;
    const int zb_nd = (ND + 255) / 256;
    const int zb_n  = (N + 255) / 256;
    const int eb    = (E + 255) / 256;
    const int sb    = (E + 7) / 8;       // 8 edges (waves) per block
    const int gb    = (N + TILE_M - 1) / TILE_M;
    const int pb    = (D * D + 255) / 256;

    // Pack weights into WMMA B-fragment layout (f16), once per call.
    sage_pack_w<<<pb, 256, 0, stream>>>(wl[0], pw[0]);
    sage_pack_w<<<pb, 256, 0, stream>>>(wr[0], pw[1]);
    sage_pack_w<<<pb, 256, 0, stream>>>(wl[1], pw[2]);
    sage_pack_w<<<pb, 256, 0, stream>>>(wr[1], pw[3]);
    sage_pack_w<<<pb, 256, 0, stream>>>(wl[2], pw[4]);
    sage_pack_w<<<pb, 256, 0, stream>>>(wr[2], pw[5]);

    // Degree (shared by all layers), then invert in place.
    sage_zero_f32<<<zb_n, 256, 0, stream>>>(deg, N);
    sage_degree<<<eb, 256, 0, stream>>>(ei + E, deg, E);
    sage_invdeg<<<zb_n, 256, 0, stream>>>(deg, N);

    const float* hin[3]  = { x, h1, h2 };
    float*       hout[3] = { h1, h2, (float*)d_out };

    for (int l = 0; l < 3; ++l) {
        sage_zero_f32<<<zb_nd, 256, 0, stream>>>(agg, ND);
        sage_scatter<<<sb, 256, 0, stream>>>(hin[l], ei, agg, E);
        sage_gemm<<<gb, 256, 0, stream>>>(hin[l], agg, deg,
                                          pw[2 * l], pw[2 * l + 1], bb[l],
                                          hout[l], N, l < 2 ? 1 : 0);
    }
}
